// Bert4KGModel_66443144069134
// MI455X (gfx1250) — matrix operand order repaired and checked by reading, not compiled, not run on hardware
//
#include <hip/hip_runtime.h>

// ---------------- model constants ----------------
#define Bc 4
#define Tc 512
#define Sc 512
#define Dc 1024
#define Hc 16
#define DHc 64
#define Fc 4096
#define Lc 2
#define LN_EPS 1e-5f
#define NEGINF -1e20f

typedef __attribute__((ext_vector_type(16))) __bf16 v16bf;
typedef __attribute__((ext_vector_type(8)))  float v8f;

union AFrag { v16bf v; uint4 q[2]; };     // 32 bytes = 2 x b128 LDS loads
union CFrag { v8f   v; float  e[8]; };

// pack two floats into a bf16x2 dword
static __device__ __forceinline__ unsigned pack2(float a, float b) {
  union { __bf16 h[2]; unsigned u; } p;
  p.h[0] = (__bf16)a; p.h[1] = (__bf16)b;
  return p.u;
}

// =====================================================================
// Embedding gather + scale + pos + LayerNorm.  One block (256 thr) per row.
// =====================================================================
__global__ void k_embed(const float* __restrict__ emb, const int* __restrict__ pv,
                        const float* __restrict__ pos, float* __restrict__ X) {
  const int row = blockIdx.x;            // b*T + t
  const int t   = row % Tc;
  const int idx = pv[row];
  const float* er = emb + (long long)idx * Dc;
  const float* pr = pos + (long long)t * Dc;
  float* xr = X + (long long)row * Dc;

  float vals[4], s = 0.f, s2 = 0.f;
#pragma unroll
  for (int i = 0; i < 4; ++i) {
    int c = threadIdx.x + (i << 8);
    float v = er[c] * 32.0f + pr[c];     // sqrt(D)=32
    vals[i] = v; s += v; s2 += v * v;
  }
  __shared__ float rs[256], rq[256];
  rs[threadIdx.x] = s; rq[threadIdx.x] = s2;
  __syncthreads();
  for (int off = 128; off; off >>= 1) {
    if (threadIdx.x < off) { rs[threadIdx.x] += rs[threadIdx.x + off];
                             rq[threadIdx.x] += rq[threadIdx.x + off]; }
    __syncthreads();
  }
  float mean = rs[0] * (1.0f / Dc);
  float var  = rq[0] * (1.0f / Dc) - mean * mean;
  float inv  = rsqrtf(var + LN_EPS);
#pragma unroll
  for (int i = 0; i < 4; ++i) {
    int c = threadIdx.x + (i << 8);
    xr[c] = (vals[i] - mean) * inv;
  }
}

// =====================================================================
// Residual add + LayerNorm.  One block (256 thr) per row of D=1024.
// =====================================================================
__global__ void k_add_ln(const float* __restrict__ x, const float* __restrict__ d,
                         float* __restrict__ out) {
  const int row = blockIdx.x;
  const float* xr = x + (long long)row * Dc;
  const float* dr = d + (long long)row * Dc;
  float* orow = out + (long long)row * Dc;

  float vals[4], s = 0.f, s2 = 0.f;
#pragma unroll
  for (int i = 0; i < 4; ++i) {
    int c = threadIdx.x + (i << 8);
    float v = xr[c] + dr[c];
    vals[i] = v; s += v; s2 += v * v;
  }
  __shared__ float rs[256], rq[256];
  rs[threadIdx.x] = s; rq[threadIdx.x] = s2;
  __syncthreads();
  for (int off = 128; off; off >>= 1) {
    if (threadIdx.x < off) { rs[threadIdx.x] += rs[threadIdx.x + off];
                             rq[threadIdx.x] += rq[threadIdx.x + off]; }
    __syncthreads();
  }
  float mean = rs[0] * (1.0f / Dc);
  float var  = rq[0] * (1.0f / Dc) - mean * mean;
  float inv  = rsqrtf(var + LN_EPS);
#pragma unroll
  for (int i = 0; i < 4; ++i) {
    int c = threadIdx.x + (i << 8);
    orow[c] = (vals[i] - mean) * inv;
  }
}

// =====================================================================
// Masked softmax over rows of the score tensor [B,H,T,S].
// mode==0: causal (s <= t); mode==1: context_mask[b,s] != cmpval.
// Applies 1/sqrt(dh) = 0.125 scaling before masking.
// =====================================================================
__global__ void k_softmax(float* __restrict__ sc, const int* __restrict__ ctx,
                          int mode, int cmpval) {
  const int row = blockIdx.x;               // (b*H + h)*T + t
  const int t = row % Tc;
  const int b = row / (Hc * Tc);
  float* p = sc + (long long)row * Sc;

  __shared__ float red[256];
  float lmax = -3.4e38f;
  for (int s = threadIdx.x; s < Sc; s += 256) {
    bool valid = (mode == 0) ? (s <= t) : (ctx[b * Sc + s] != cmpval);
    float v = valid ? p[s] * 0.125f : NEGINF;
    p[s] = v;
    lmax = fmaxf(lmax, v);
  }
  red[threadIdx.x] = lmax; __syncthreads();
  for (int off = 128; off; off >>= 1) {
    if (threadIdx.x < off) red[threadIdx.x] = fmaxf(red[threadIdx.x], red[threadIdx.x + off]);
    __syncthreads();
  }
  float rmax = red[0]; __syncthreads();

  float lsum = 0.f;
  for (int s = threadIdx.x; s < Sc; s += 256) {
    float e = __expf(p[s] - rmax);
    p[s] = e; lsum += e;
  }
  red[threadIdx.x] = lsum; __syncthreads();
  for (int off = 128; off; off >>= 1) {
    if (threadIdx.x < off) red[threadIdx.x] += red[threadIdx.x + off];
    __syncthreads();
  }
  float inv = 1.0f / red[0];
  for (int s = threadIdx.x; s < Sc; s += 256) p[s] *= inv;
}

// =====================================================================
// Generalized batched GEMM:  C[z] = act( A[z][MxK] * B[z][KxN] + bias )
// fp32 operands converted to bf16 (packed bf16x2 dwords) while staging to
// LDS; f32 accumulation via v_wmma_f32_16x16x32_bf16.
//   Block tile 128x128, BK=32, 256 threads (8 wave32s);
//   each wave computes a 32x64 patch = 2x4 WMMA tiles.
// LDS layouts are fragment-native: each 16-bf16 fragment = two 16B-aligned
// ds_load_b128 ops.  Compile-time specialization <TRANSB, BOUNDED> removes
// all per-iteration branch dispatch:
//   <0,0>: dense projections / FFN (M,N multiples of 128) - no guards
//   <1,0>: Q @ K^T scores - transposed-B staging, no guards
//   <0,1>: P @ V (N=64) - per-element N guards
// z = (zb, zh): per-batch/per-head stride offsets (zb = z/Hdiv, zh = z%Hdiv).
// =====================================================================
template <int TRANSB, int BOUNDED>
__global__ __launch_bounds__(256)
void k_gemm(const float* __restrict__ A, const float* __restrict__ Bsrc,
            const float* __restrict__ bias, float* __restrict__ C,
            int M, int N, int K, int lda, int ldb, int ldc,
            long long sAb, long long sAh, long long sBb, long long sBh,
            long long sCb, long long sCh,
            int Hdiv, int doRelu) {
  constexpr int BM = 128, BN = 128, BK = 32;
  constexpr int AP = 20;   // dword units per A row (16 kpairs + 4 pad)
  constexpr int BP = 20;   // dword units per B column (16 kpairs + 4 pad)

  __shared__ unsigned Asu[2][BM * AP];   // 20 KB
  __shared__ unsigned Bsu[2][BN * BP];   // 20 KB

  const int z  = blockIdx.z;
  const int zb = z / Hdiv;
  const int zh = z % Hdiv;
  const long long baseA = (long long)zb * sAb + (long long)zh * sAh;
  const long long baseB = (long long)zb * sBb + (long long)zh * sBh;
  const long long baseC = (long long)zb * sCb + (long long)zh * sCh;

  const int m0 = blockIdx.y * BM;
  const int n0 = blockIdx.x * BN;
  const int tid = (int)threadIdx.x;

  const int lane  = tid & 31;
  const int wave  = tid >> 5;
  const int waveM = wave & 3;    // 4 waves along M -> 4*32 = 128
  const int waveN = wave >> 2;   // 2 waves along N -> 2*64 = 128
  const int half  = lane >> 4;   // hi/lo 16-lane group (ISA fragment layout)
  const int l15   = lane & 15;

  // ---- global (fp32) -> LDS (packed bf16x2) stagers ----------------------
  // Two-phase: issue all global loads back-to-back, then convert + store.
  auto loadA = [&](int k0, int buf) {
    float4 ra[4];
#pragma unroll
    for (int i = 0; i < 4; ++i) {
      const int s = tid + (i << 8);
      const int r = s >> 3, c = (s & 7) << 2;
      ra[i] = *(const float4*)(A + baseA + (long long)(m0 + r) * lda + (k0 + c));
    }
#pragma unroll
    for (int i = 0; i < 4; ++i) {
      const int s = tid + (i << 8);
      const int r = s >> 3, c = (s & 7) << 2;
      uint2 u; u.x = pack2(ra[i].x, ra[i].y); u.y = pack2(ra[i].z, ra[i].w);
      *(uint2*)&Asu[buf][r * AP + (c >> 1)] = u;
    }
  };

  auto loadB = [&](int k0, int buf) {
    if constexpr (TRANSB) {
      // B element (k,n) = Bsrc[n*ldb + k] (Q @ K^T): contiguous along k.
      float4 rb[4];
#pragma unroll
      for (int i = 0; i < 4; ++i) {
        const int s = tid + (i << 8);
        const int nn = s >> 3, kc = (s & 7) << 2;
        rb[i] = *(const float4*)(Bsrc + baseB + (long long)(n0 + nn) * ldb + (k0 + kc));
      }
#pragma unroll
      for (int i = 0; i < 4; ++i) {
        const int s = tid + (i << 8);
        const int nn = s >> 3, kc = (s & 7) << 2;
        uint2 u; u.x = pack2(rb[i].x, rb[i].y); u.y = pack2(rb[i].z, rb[i].w);
        *(uint2*)&Bsu[buf][nn * BP + (kc >> 1)] = u;
      }
    } else if constexpr (!BOUNDED) {
      // B row-major [K x N]: two adjacent k-rows -> packed bf16x2 units.
      float4 rb0[2], rb1[2];
#pragma unroll
      for (int i = 0; i < 2; ++i) {
        const int s = tid + (i << 8);
        const int kp = s >> 5, c = (s & 31) << 2;
        const float* r0 = Bsrc + baseB + (long long)(k0 + 2 * kp) * ldb + (n0 + c);
        rb0[i] = *(const float4*)r0;
        rb1[i] = *(const float4*)(r0 + ldb);
      }
#pragma unroll
      for (int i = 0; i < 2; ++i) {
        const int s = tid + (i << 8);
        const int kp = s >> 5, c = (s & 31) << 2;
        Bsu[buf][(c + 0) * BP + kp] = pack2(rb0[i].x, rb1[i].x);
        Bsu[buf][(c + 1) * BP + kp] = pack2(rb0[i].y, rb1[i].y);
        Bsu[buf][(c + 2) * BP + kp] = pack2(rb0[i].z, rb1[i].z);
        Bsu[buf][(c + 3) * BP + kp] = pack2(rb0[i].w, rb1[i].w);
      }
    } else {
      // Bounded (P @ V, N=64): per-element column guards.
#pragma unroll
      for (int i = 0; i < 2; ++i) {
        const int s = tid + (i << 8);
        const int kp = s >> 5, c = (s & 31) << 2;
        const int gn = n0 + c;
        const float* r0 = Bsrc + baseB + (long long)(k0 + 2 * kp) * ldb + gn;
        const float* r1 = r0 + ldb;
        float a0 = (gn + 0 < N) ? r0[0] : 0.f, b0 = (gn + 0 < N) ? r1[0] : 0.f;
        float a1 = (gn + 1 < N) ? r0[1] : 0.f, b1 = (gn + 1 < N) ? r1[1] : 0.f;
        float a2 = (gn + 2 < N) ? r0[2] : 0.f, b2 = (gn + 2 < N) ? r1[2] : 0.f;
        float a3 = (gn + 3 < N) ? r0[3] : 0.f, b3 = (gn + 3 < N) ? r1[3] : 0.f;
        Bsu[buf][(c + 0) * BP + kp] = pack2(a0, b0);
        Bsu[buf][(c + 1) * BP + kp] = pack2(a1, b1);
        Bsu[buf][(c + 2) * BP + kp] = pack2(a2, b2);
        Bsu[buf][(c + 3) * BP + kp] = pack2(a3, b3);
      }
    }
  };

  v8f acc[2][4] = {};

  const int nk = K / BK;          // K is always a multiple of 32 here
  int buf = 0;
  loadA(0, 0);
  loadB(0, 0);
  __syncthreads();

  for (int ks = 0; ks < nk; ++ks) {
    const int nxt = buf ^ 1;
    if (ks + 1 < nk) {            // double-buffered staging into other LDS bank
      loadA((ks + 1) * BK, nxt);
      loadB((ks + 1) * BK, nxt);
      if (ks + 2 < nk) {          // L2 prefetch hint for tile after next
        __builtin_prefetch(A + baseA + (long long)(m0 + (tid & 127)) * lda
                             + (ks + 2) * BK, 0, 0);
      }
    }

    // ---- fragment loads: 2 x b128 per fragment (ISA 7.12.2 layouts) -----
    AFrag af[2];
#pragma unroll
    for (int mt = 0; mt < 2; ++mt) {
      const int row = waveM * 32 + mt * 16 + l15;
      const unsigned* ap = &Asu[buf][row * AP + half * 4];
      af[mt].q[0] = *(const uint4*)(ap);      // K = half*8 + 0..7
      af[mt].q[1] = *(const uint4*)(ap + 8);  // K = 16 + half*8 + 0..7
    }
    AFrag bfr[4];
#pragma unroll
    for (int nt = 0; nt < 4; ++nt) {
      const int col = waveN * 64 + nt * 16 + l15;
      const unsigned* bp = &Bsu[buf][col * BP + half * 4];
      bfr[nt].q[0] = *(const uint4*)(bp);     // K = half*8 + 0..7
      bfr[nt].q[1] = *(const uint4*)(bp + 8); // K = 16 + half*8 + 0..7
    }

#pragma unroll
    for (int mt = 0; mt < 2; ++mt)
#pragma unroll
      for (int nt = 0; nt < 4; ++nt)
        acc[mt][nt] = __builtin_amdgcn_wmma_f32_16x16x32_bf16(
            false, af[mt].v, false, bfr[nt].v, (short)0, acc[mt][nt], false, false);

    __syncthreads();
    buf = nxt;
  }

  // ---- epilogue: bias (+ReLU) + store ------------------------------------
#pragma unroll
  for (int mt = 0; mt < 2; ++mt) {
#pragma unroll
    for (int nt = 0; nt < 4; ++nt) {
      CFrag c; c.v = acc[mt][nt];
      const int col = n0 + waveN * 64 + nt * 16 + l15;
      if (!BOUNDED || col < N) {
        const float bv = bias ? bias[col] : 0.f;
#pragma unroll
        for (int r = 0; r < 8; ++r) {
          // C/D layout: VGPR r -> M = r + 8*half, N = lane&15
          const int rowg = m0 + waveM * 32 + mt * 16 + half * 8 + r;
          float x = c.e[r] + bv;
          if (doRelu) x = fmaxf(x, 0.f);
          C[baseC + (long long)rowg * ldc + col] = x;
        }
      }
    }
  }
}

// =====================================================================
// Host-side orchestration
// =====================================================================
template <int TRANSB, int BOUNDED>
static inline void gemm(hipStream_t st, const float* A, const float* Bp,
                        const float* bias, float* C,
                        int M, int N, int K, int lda, int ldb, int ldc,
                        long long sAb, long long sAh, long long sBb, long long sBh,
                        long long sCb, long long sCh,
                        int Z, int Hdiv, int doRelu) {
  dim3 grid((N + 127) / 128, (M + 127) / 128, Z);
  k_gemm<TRANSB, BOUNDED><<<grid, 256, 0, st>>>(
      A, Bp, bias, C, M, N, K, lda, ldb, ldc,
      sAb, sAh, sBb, sBh, sCb, sCh, Hdiv, doRelu);
}

extern "C" void kernel_launch(void* const* d_in, const int* in_sizes, int n_in,
                              void* d_out, int out_size, void* d_ws, size_t ws_size,
                              hipStream_t stream) {
  (void)in_sizes; (void)n_in; (void)out_size; (void)ws_size;

  const float* emb  = (const float*)d_in[0];
  const int*   pv   = (const int*)  d_in[1];
  const float* enc  = (const float*)d_in[2];
  const float* con  = (const float*)d_in[3];
  const float* db   = (const float*)d_in[4];
  const float* usr  = (const float*)d_in[5];
  const int*   ctx  = (const int*)  d_in[6];
  const float* pos  = (const float*)d_in[7];
  const float* Wq = (const float*)d_in[8],  *bq = (const float*)d_in[9];
  const float* Wk = (const float*)d_in[10], *bk = (const float*)d_in[11];
  const float* Wv = (const float*)d_in[12], *bv = (const float*)d_in[13];
  const float* Wo = (const float*)d_in[14], *bo = (const float*)d_in[15];
  const float* W1 = (const float*)d_in[16], *b1 = (const float*)d_in[17];
  const float* W2 = (const float*)d_in[18], *b2 = (const float*)d_in[19];
  float* out = (float*)d_out;

  // workspace bump allocator (all fp32, 256B aligned)
  char* w = (char*)d_ws;
  auto alloc = [&](long long elems) {
    float* p = (float*)w;
    w += ((elems * 4 + 255) / 256) * 256;
    return p;
  };
  const long long MT = (long long)Bc * Tc;         // 2048 token rows
  float* xbuf = alloc(MT * Dc);
  float* qbuf = alloc(MT * Dc);
  float* kbuf = alloc(MT * Dc);
  float* vbuf = alloc(MT * Dc);
  float* obuf = alloc(MT * Dc);
  float* pbuf = alloc(MT * Dc);
  float* sbuf = alloc((long long)Bc * Hc * Tc * Sc);
  float* hbuf = alloc(MT * Fc);

  // embedding + pos + LN
  k_embed<<<(int)MT, 256, 0, stream>>>(emb, pv, pos, xbuf);

  const float* kvsrc[5] = {nullptr /*self: x*/, db, con, usr, enc};
  const int    mmode[5] = {0, 1, 1, 1, 1};            // 0 = causal
  const int    mval [5] = {0, 2 /*DBPEDIA*/, 1 /*CONCEPT*/, 3 /*USER*/, 0 /*PAD*/};

  const long long DD = (long long)Dc * Dc;
  const long long ZTS = (long long)Tc * Sc;

  for (int l = 0; l < Lc; ++l) {
    for (int i = 0; i < 5; ++i) {
      const float* kv = (i == 0) ? xbuf : kvsrc[i];
      const long long wi = (long long)(l * 5 + i);

      // Q/K/V projections: [2048,1024] x [1024,1024]
      gemm<0, 0>(stream, xbuf, Wq + wi * DD, bq + wi * Dc, qbuf,
                 (int)MT, Dc, Dc, Dc, Dc, Dc, 0, 0, 0, 0, 0, 0, 1, 1, 0);
      gemm<0, 0>(stream, kv, Wk + wi * DD, bk + wi * Dc, kbuf,
                 (int)MT, Dc, Dc, Dc, Dc, Dc, 0, 0, 0, 0, 0, 0, 1, 1, 0);
      gemm<0, 0>(stream, kv, Wv + wi * DD, bv + wi * Dc, vbuf,
                 (int)MT, Dc, Dc, Dc, Dc, Dc, 0, 0, 0, 0, 0, 0, 1, 1, 0);

      // scores[z=(b,h)] = Q_bh [T,dh] x K_bh^T [dh,S]
      gemm<1, 0>(stream, qbuf, kbuf, nullptr, sbuf,
                 Tc, Sc, DHc, Dc, Dc, Sc,
                 (long long)Tc * Dc, DHc,             // A: b*T*D + h*dh
                 (long long)Sc * Dc, DHc,             // B: b*S*D + h*dh (transB)
                 (long long)Hc * ZTS, ZTS,            // C: (b*H+h)*T*S
                 Bc * Hc, Hc, 0);

      // masked softmax
      k_softmax<<<Bc * Hc * Tc, 256, 0, stream>>>(sbuf, ctx, mmode[i], mval[i]);

      // out[z] = P_bh [T,S] x V_bh [S,dh]   (N=64 -> bounded variant)
      gemm<0, 1>(stream, sbuf, vbuf, nullptr, obuf,
                 Tc, DHc, Sc, Sc, Dc, Dc,
                 (long long)Hc * ZTS, ZTS,            // A: (b*H+h)*T*S
                 (long long)Sc * Dc, DHc,             // B: b*S*D + h*dh
                 (long long)Tc * Dc, DHc,             // C: b*T*D + h*dh
                 Bc * Hc, Hc, 0);

      // output projection + residual LN
      gemm<0, 0>(stream, obuf, Wo + wi * DD, bo + wi * Dc, pbuf,
                 (int)MT, Dc, Dc, Dc, Dc, Dc, 0, 0, 0, 0, 0, 0, 1, 1, 0);
      k_add_ln<<<(int)MT, 256, 0, stream>>>(xbuf, pbuf, xbuf);
    }

    // FFN: relu(x W1 + b1) W2 + b2, residual LN (last layer writes d_out)
    gemm<0, 0>(stream, xbuf, W1 + (long long)l * Dc * Fc, b1 + (long long)l * Fc, hbuf,
               (int)MT, Fc, Dc, Dc, Fc, Fc, 0, 0, 0, 0, 0, 0, 1, 1, 1);
    gemm<0, 0>(stream, hbuf, W2 + (long long)l * Fc * Dc, b2 + (long long)l * Dc, pbuf,
               (int)MT, Dc, Fc, Fc, Dc, Dc, 0, 0, 0, 0, 0, 0, 1, 1, 0);
    float* dst = (l == Lc - 1) ? out : xbuf;
    k_add_ln<<<(int)MT, 256, 0, stream>>>(xbuf, pbuf, dst);
  }
}
